// TrajectoryDiscriminator_68590627717130
// MI455X (gfx1250) — matrix operand order repaired and checked by reading, not compiled, not run on hardware
//
#include <hip/hip_runtime.h>
#include <math.h>

// CDNA5 / gfx1250 fused TrajectoryDiscriminator forward.
// One workgroup (256 thr = 8 wave32) per batch; all recurrence state in LDS;
// edge matrices streamed from HBM once (bandwidth floor ~126MB / 23.3TB/s).

typedef _Float16 v16h __attribute__((ext_vector_type(16)));
typedef float    v8f  __attribute__((ext_vector_type(8)));

namespace {

constexpr int Bb = 128, Nn = 64, Tt = 20, Fd = 2, Ee = 16, Hh = 32;
constexpr float EPSv = 1e-12f;

// Total dynamic LDS (floats)
constexpr int SM_FLOATS =
    64*128   // Bg   : gate weights, B-layout [K=64][N=128]
  + 64*32    // Bspa : [K=64][N=32]
  + 32*32    // Be2a : [K=32][N=32]
  + 32*32    // Ba2e : [K=32][N=32]
  + 128      // bg   : b_ih + b_hh
  + 32 + 32 + 32   // be2a, bspa, ba2e
  + 32 + 16  // embW, embB
  + 144      // domain 12x12
  + 32       // clsW
  + 64*32    // hF
  + 64*32    // cF
  + 64*64    // CC   : cols 0..31 = a, 32..63 = att
  + 64*64    // Abuf : gate-A / w / z scratch
  + 64*128   // gF   : LSTM pre-activations
  + 256;     // wpart: row-sum partials

__device__ __forceinline__ float sigf(float v) { return 1.f / (1.f + __expf(-v)); }

// ---- WMMA fragment loaders (ISA 7.12.2 layouts), f32 LDS -> f16 regs ----

// A-matrix 16x32 f16: lane&15 = M row; lane>=16 selects K offset +8 within each
// 4-VGPR half; VGPR v holds K pair ( (v&4)*4 + (lane>>4)*8 + (v&3)*2 ).
__device__ __forceinline__ v16h ldsA(const float* A, int lda, int row0, int k0, int lane) {
  const int m  = lane & 15;
  const int kb = (lane >> 4) * 8;
  const float* rp = A + (row0 + m) * lda + k0;
  v16h a;
#pragma unroll
  for (int v = 0; v < 8; ++v) {
    const int k = ((v & 4) << 2) + kb + ((v & 3) << 1);
    a[2*v]     = (_Float16)rp[k];
    a[2*v + 1] = (_Float16)rp[k + 1];
  }
  return a;
}

// B-matrix 32x16 f16: lane&15 = N col; same K order as A: VGPR v holds K pair
// ( (v&4)*4 + (lane>>4)*8 + (v&3)*2 ).
__device__ __forceinline__ v16h ldsB(const float* Bm, int ldb, int k0, int col0, int lane) {
  const int n  = col0 + (lane & 15);
  const int kb = k0 + ((lane >> 4) << 3);
  v16h b;
#pragma unroll
  for (int v = 0; v < 8; ++v) {
    const int k = kb + ((v & 4) << 2) + ((v & 3) << 1);
    b[2*v]     = (_Float16)Bm[k * ldb + n];
    b[2*v + 1] = (_Float16)Bm[(k + 1) * ldb + n];
  }
  return b;
}

__device__ __forceinline__ v8f wmma(v16h a, v16h b, v8f c) {
  return __builtin_amdgcn_wmma_f32_16x16x32_f16(false, a, false, b, (short)0, c,
                                                false, false);
}

__global__ __launch_bounds__(256)
void traj_disc_kernel(const float* __restrict__ x,    const float* __restrict__ dmat,
                      const float* __restrict__ bmat, const float* __restrict__ hmat,
                      const float* __restrict__ mask, const float* __restrict__ embW,
                      const float* __restrict__ embB, const float* __restrict__ Wih,
                      const float* __restrict__ Whh,  const float* __restrict__ bih,
                      const float* __restrict__ bhh,  const float* __restrict__ domain,
                      const float* __restrict__ e2aW, const float* __restrict__ e2aB,
                      const float* __restrict__ spaW, const float* __restrict__ spaB,
                      const float* __restrict__ a2eW, const float* __restrict__ a2eB,
                      const float* __restrict__ clsW, const float* __restrict__ clsB,
                      const float* __restrict__ h0,   const float* __restrict__ c0,
                      float* __restrict__ out) {
  extern __shared__ float sm[];
  float* Bg    = sm;                 // [64][128]
  float* Bspa  = Bg    + 64 * 128;   // [64][32]
  float* Be2a  = Bspa  + 64 * 32;    // [32][32]
  float* Ba2e  = Be2a  + 32 * 32;    // [32][32]
  float* bg    = Ba2e  + 32 * 32;    // [128]
  float* be2a  = bg    + 128;        // [32]
  float* bspa  = be2a  + 32;         // [32]
  float* ba2e  = bspa  + 32;         // [32]
  float* embWs = ba2e  + 32;         // [32]  (E x F = 16x2)
  float* embBs = embWs + 32;         // [16]
  float* dom   = embBs + 16;         // [144]
  float* clsWs = dom   + 144;        // [32]
  float* hF    = clsWs + 32;         // [64][32]
  float* cF    = hF    + 64 * 32;    // [64][32]
  float* CC    = cF    + 64 * 32;    // [64][64]
  float* Abuf  = CC    + 64 * 64;    // [64][64]
  float* gF    = Abuf  + 64 * 64;    // [64][128]
  float* wpart = gF    + 64 * 128;   // [256]

  const int b    = blockIdx.x;
  const int tid  = threadIdx.x;
  const int lane = tid & 31;
  const int wv   = tid >> 5;  // wave id 0..7

  // ---------------- one-time weight staging (B-matrix = W^T) --------------
  for (int idx = tid; idx < 64 * 128; idx += 256) {
    const int k = idx >> 7, n = idx & 127;
    float v = 0.f;
    if (k < 16)      v = Wih[n * 16 + k];          // W_ih [128][16]
    else if (k < 48) v = Whh[n * 32 + (k - 16)];   // W_hh [128][32]
    Bg[idx] = v;                                   // K pad 48..63 = 0
  }
  for (int idx = tid; idx < 64 * 32; idx += 256) {
    const int k = idx >> 5, n = idx & 31;
    Bspa[idx] = spaW[n * 64 + k];                  // spa_W [32][64]
  }
  for (int idx = tid; idx < 32 * 32; idx += 256) {
    const int k = idx >> 5, n = idx & 31;
    Be2a[idx] = e2aW[n * 32 + k];
    Ba2e[idx] = a2eW[n * 32 + k];
  }
  if (tid < 128) bg[tid] = bih[tid] + bhh[tid];
  if (tid < 32) {
    be2a[tid]  = e2aB[tid];
    bspa[tid]  = spaB[tid];
    ba2e[tid]  = a2eB[tid];
    embWs[tid] = embW[tid];
    clsWs[tid] = clsW[tid];
  }
  if (tid < 16)  embBs[tid] = embB[tid];
  if (tid < 144) dom[tid]   = domain[tid];
  for (int idx = tid; idx < 64 * 32; idx += 256) {   // init recurrent state
    hF[idx] = h0[(size_t)b * 64 * 32 + idx];
    cF[idx] = c0[(size_t)b * 64 * 32 + idx];
  }
  __syncthreads();

  // ---------------------------- time recurrence ---------------------------
  for (int t = 0; t < Tt; ++t) {
    // s0: gate A-matrix  Abuf[i][0:16]=xe, [16:48]=h, [48:64]=0
    for (int idx = tid; idx < 64 * 64; idx += 256) {
      const int i = idx >> 6, k = idx & 63;
      float v;
      if (k < 16) {
        const float* xp = x + (((size_t)(b * 64 + i) * Tt + t) * Fd);
        v = xp[0] * embWs[k * 2] + xp[1] * embWs[k * 2 + 1] + embBs[k];
      } else if (k < 48) {
        v = hF[i * 32 + (k - 16)];
      } else {
        v = 0.f;
      }
      Abuf[idx] = v;
    }
    __syncthreads();

    // s1: gates g[64x128] = A[64x64] @ Bg[64x128] + bias  (32 tiles, 4/wave)
#pragma unroll
    for (int tt = 0; tt < 4; ++tt) {
      const int tile = wv * 4 + tt;
      const int mt = tile >> 3, nt = tile & 7;
      v8f acc = {};
      acc = wmma(ldsA(Abuf, 64, mt * 16, 0,  lane), ldsB(Bg, 128, 0,  nt * 16, lane), acc);
      acc = wmma(ldsA(Abuf, 64, mt * 16, 32, lane), ldsB(Bg, 128, 32, nt * 16, lane), acc);
      const int nn = nt * 16 + (lane & 15);
      const int mb = mt * 16 + ((lane >> 4) << 3);
#pragma unroll
      for (int v = 0; v < 8; ++v) gF[(mb + v) * 128 + nn] = acc[v] + bg[nn];
    }
    __syncthreads();

    // s2a: LSTM pointwise (torch gate order i,f,g,o)
#pragma unroll
    for (int r = 0; r < 8; ++r) {
      const int e = tid * 8 + r;
      const int i = e >> 5, j = e & 31;
      const float ig = sigf(gF[i * 128 + j]);
      const float fg = sigf(gF[i * 128 + 32 + j]);
      const float gg = tanhf(gF[i * 128 + 64 + j]);
      const float og = sigf(gF[i * 128 + 96 + j]);
      const float cc = fg * cF[e] + ig * gg;
      cF[e] = cc;
      hF[e] = og * tanhf(cc);
    }
    // s2b: spatial weights w -> Abuf (streams dmat/bmat/hmat from HBM once)
    {
      const int    i     = tid >> 2;
      const int    jq    = (tid & 3) << 4;
      const size_t ebase = ((size_t)(b * 64 + i) * Tt + t) * 64;
      const float  mki   = mask[(size_t)(b * 64 + i) * Tt + t];
      float psum = 0.f;
#pragma unroll
      for (int jj = 0; jj < 16; ++jj) {
        const int j = jq + jj;
        const float dv = dmat[ebase + j];
        const float bv = bmat[ebase + j];
        const float hv = hmat[ebase + j];
        int ib = (int)floorf(bv * (1.f / 30.f)); ib = ib < 0 ? 0 : (ib > 11 ? 11 : ib);
        int ih = (int)floorf(hv * (1.f / 30.f)); ih = ih < 0 ? 0 : (ih > 11 ? 11 : ih);
        float wvv = dom[ib * 12 + ih] - dv;
        wvv = wvv > 0.f ? wvv : 0.f;
        const float mkj = mask[(size_t)(b * 64 + j) * Tt + t];
        wvv *= mki * mkj * (i == j ? 0.f : 1.f);
        Abuf[i * 64 + j] = wvv;
        psum += wvv;
      }
      if (t + 1 < Tt) {  // next step is +64 floats in the [B,N,T,N] layout
        __builtin_prefetch(dmat + ebase + 64 + jq, 0, 1);
        __builtin_prefetch(bmat + ebase + 64 + jq, 0, 1);
        __builtin_prefetch(hmat + ebase + 64 + jq, 0, 1);
      }
      wpart[tid] = psum;
    }
    __syncthreads();
    {  // s2c: row-normalize w in place
      const int i  = tid >> 2;
      const int jq = (tid & 3) << 4;
      const float rs  = wpart[i * 4] + wpart[i * 4 + 1] + wpart[i * 4 + 2] + wpart[i * 4 + 3];
      const float inv = 1.f / (rs + EPSv);
#pragma unroll
      for (int jj = 0; jj < 16; ++jj) Abuf[i * 64 + jq + jj] *= inv;
    }
    __syncthreads();

    // s3: a = h @ e2aW^T + b  -> CC[:,0:32]   (8 tiles, 1/wave)
    {
      const int mt = wv >> 1, nt = wv & 1;
      v8f acc = {};
      acc = wmma(ldsA(hF, 32, mt * 16, 0, lane), ldsB(Be2a, 32, 0, nt * 16, lane), acc);
      const int nn = nt * 16 + (lane & 15);
      const int mb = mt * 16 + ((lane >> 4) << 3);
#pragma unroll
      for (int v = 0; v < 8; ++v) CC[(mb + v) * 64 + nn] = acc[v] + be2a[nn];
    }
    __syncthreads();

    // s4: att = w @ a -> CC[:,32:64]
    {
      const int mt = wv >> 1, nt = wv & 1;
      v8f acc = {};
      acc = wmma(ldsA(Abuf, 64, mt * 16, 0,  lane), ldsB(CC, 64, 0,  nt * 16, lane), acc);
      acc = wmma(ldsA(Abuf, 64, mt * 16, 32, lane), ldsB(CC, 64, 32, nt * 16, lane), acc);
      const int nn = nt * 16 + (lane & 15);
      const int mb = mt * 16 + ((lane >> 4) << 3);
#pragma unroll
      for (int v = 0; v < 8; ++v) CC[(mb + v) * 64 + 32 + nn] = acc[v];
    }
    __syncthreads();

    // s5: z = tanh([a|att] @ spaW^T + b) -> Abuf[:,0:32]
    {
      const int mt = wv >> 1, nt = wv & 1;
      v8f acc = {};
      acc = wmma(ldsA(CC, 64, mt * 16, 0,  lane), ldsB(Bspa, 32, 0,  nt * 16, lane), acc);
      acc = wmma(ldsA(CC, 64, mt * 16, 32, lane), ldsB(Bspa, 32, 32, nt * 16, lane), acc);
      const int nn = nt * 16 + (lane & 15);
      const int mb = mt * 16 + ((lane >> 4) << 3);
#pragma unroll
      for (int v = 0; v < 8; ++v) Abuf[(mb + v) * 64 + nn] = tanhf(acc[v] + bspa[nn]);
    }
    __syncthreads();

    // s6: h = z @ a2eW^T + b -> hF (next-step recurrent h)
    {
      const int mt = wv >> 1, nt = wv & 1;
      v8f acc = {};
      acc = wmma(ldsA(Abuf, 64, mt * 16, 0, lane), ldsB(Ba2e, 32, 0, nt * 16, lane), acc);
      const int nn = nt * 16 + (lane & 15);
      const int mb = mt * 16 + ((lane >> 4) << 3);
#pragma unroll
      for (int v = 0; v < 8; ++v) hF[(mb + v) * 32 + nn] = acc[v] + ba2e[nn];
    }
    __syncthreads();
  }

  // classifier: sigmoid(tanh(h @ cls_W^T + b))
  if (tid < 64) {
    float s = clsB[0];
#pragma unroll
    for (int j = 0; j < 32; ++j) s += hF[tid * 32 + j] * clsWs[j];
    s = tanhf(s);
    out[(size_t)b * 64 + tid] = 1.f / (1.f + __expf(-s));
  }
}

}  // namespace

extern "C" void kernel_launch(void* const* d_in, const int* in_sizes, int n_in,
                              void* d_out, int out_size, void* d_ws, size_t ws_size,
                              hipStream_t stream) {
  (void)in_sizes; (void)n_in; (void)out_size; (void)d_ws; (void)ws_size;
  const float* x     = (const float*)d_in[0];
  const float* dmat  = (const float*)d_in[1];
  const float* bmat  = (const float*)d_in[2];
  const float* hmat  = (const float*)d_in[3];
  const float* mask  = (const float*)d_in[4];
  const float* embW  = (const float*)d_in[5];
  const float* embB  = (const float*)d_in[6];
  const float* Wih   = (const float*)d_in[7];
  const float* Whh   = (const float*)d_in[8];
  const float* bih   = (const float*)d_in[9];
  const float* bhh   = (const float*)d_in[10];
  const float* dom   = (const float*)d_in[11];
  const float* e2aW  = (const float*)d_in[12];
  const float* e2aB  = (const float*)d_in[13];
  const float* spaW  = (const float*)d_in[14];
  const float* spaB  = (const float*)d_in[15];
  const float* a2eW  = (const float*)d_in[16];
  const float* a2eB  = (const float*)d_in[17];
  const float* clsW  = (const float*)d_in[18];
  const float* clsB  = (const float*)d_in[19];
  const float* h0    = (const float*)d_in[20];
  const float* c0    = (const float*)d_in[21];
  float* out = (float*)d_out;

  const size_t smem = (size_t)SM_FLOATS * sizeof(float);  // ~131 KB < 320 KB WGP LDS
  traj_disc_kernel<<<dim3(Bb), dim3(256), smem, stream>>>(
      x, dmat, bmat, hmat, mask, embW, embB, Wih, Whh, bih, bhh, dom,
      e2aW, e2aB, spaW, spaB, a2eW, a2eB, clsW, clsB, h0, c0, out);
}
